// Attention_7284264534637
// MI455X (gfx1250) — matrix operand order repaired and checked by reading, not compiled, not run on hardware
//
#include <hip/hip_runtime.h>
#include <hip/hip_bf16.h>
#include <math.h>

typedef __attribute__((ext_vector_type(16))) __bf16 v16bf;
typedef __attribute__((ext_vector_type(8)))  __bf16 v8bf;
typedef __attribute__((ext_vector_type(8)))  float  v8f;

#define B_   32
#define S_   2048
#define DEC_ 512
#define ENC_ 1024
#define WROW 1536   // DEC_ + ENC_

#if __has_builtin(__builtin_amdgcn_tanhf)
#define TANHF(x) __builtin_amdgcn_tanhf(x)
#else
#define TANHF(x) tanhf(x)
#endif

// XOR-butterfly within 16-lane halves via ds_swizzle (and=0x1F, or=0, xor=imm<<10)
#if __has_builtin(__builtin_amdgcn_ds_swizzle)
#define SWZ_ADD(t, imm) \
    t += __int_as_float(__builtin_amdgcn_ds_swizzle(__float_as_int(t), imm))
#define RED16(t) { SWZ_ADD(t, 0x041F); SWZ_ADD(t, 0x081F); \
                   SWZ_ADD(t, 0x101F); SWZ_ADD(t, 0x201F); }
#else
#define RED16(t) { t += __shfl_xor(t, 1, 32); t += __shfl_xor(t, 2, 32); \
                   t += __shfl_xor(t, 4, 32); t += __shfl_xor(t, 8, 32); }
#endif

// ---------------------------------------------------------------------------
// Workspace layout (bytes):
//   [0, 1MB)                : web   bf16[512*1024]   (W_e converted)
//   [1MB, 1MB+64KB)         : sW    f32 [32*512]     (s @ W_s^T)
//   [1MB+64KB, +256KB)      : atten f32 [65536]      (pre-softmax logits)
// ---------------------------------------------------------------------------

__global__ __launch_bounds__(256) void k_convW(const float* __restrict__ Wattn,
                                               __bf16* __restrict__ web) {
    int idx = blockIdx.x * 256 + threadIdx.x;      // 0 .. 512*1024
    int h = idx >> 10, k = idx & 1023;
    web[idx] = (__bf16)Wattn[(size_t)h * WROW + DEC_ + k];
}

__global__ __launch_bounds__(256) void k_sW(const float* __restrict__ s,
                                            const float* __restrict__ Wattn,
                                            float* __restrict__ sW) {
    int idx = blockIdx.x * 256 + threadIdx.x;      // 0 .. 32*512
    int b = idx >> 9, h = idx & 511;
    const float4* sp = (const float4*)(s + (size_t)b * DEC_);
    const float4* wp = (const float4*)(Wattn + (size_t)h * WROW);
    float acc = 0.f;
    for (int i = 0; i < DEC_ / 4; ++i) {
        float4 a = sp[i], w = wp[i];
        acc += a.x * w.x + a.y * w.y + a.z * w.z + a.w * w.w;
    }
    sW[idx] = acc;
}

// Main fused kernel: E = enc @ W_e^T (bf16 WMMA, fp32 accum), then
// atten[row] = sum_h W_v[h] * tanh(sW[b,h] + E[row,h]).
// WG = 8 waves, covers 32 rows (two 16-row M tiles) x all 512 h.
// Each wave: 4 N-tiles x 2 M-tiles = 8 v8f accumulators; every B fragment
// feeds two WMMAs (halves W_e L2 traffic vs one M tile per wave).
__global__ __launch_bounds__(256) void k_energy(const float* __restrict__ enc,
                                                const __bf16* __restrict__ web,
                                                const float* __restrict__ sW,
                                                const float* __restrict__ Wv,
                                                float* __restrict__ atten) {
    __shared__ float lds_atten[32];
    const int tid  = threadIdx.x;
    const int lane = tid & 31;
    const int wid  = tid >> 5;        // wave 0..7, owns n-tiles wid*4 .. wid*4+3
    const int half = lane >> 4;       // 0 or 1 (half-wave)
    const int m16  = lane & 15;
    const int m0   = blockIdx.x * 32; // 32-row M block of flattened (b,s)
    const int b    = m0 >> 11;        // 32 | 2048 -> block never crosses batches

    if (tid < 32) lds_atten[tid] = 0.f;
    __syncthreads();

    v8f acc00 = {}, acc01 = {}, acc02 = {}, acc03 = {};   // M tile 0
    v8f acc10 = {}, acc11 = {}, acc12 = {}, acc13 = {};   // M tile 1

    // A: fp32 enc rows for this lane's two M indices (bf16 on the fly)
    const float* arow0 = enc + (size_t)(m0 + m16) * ENC_;
    const float* arow1 = enc + (size_t)(m0 + 16 + m16) * ENC_;
    // B: pre-converted bf16 W_e rows; h = (wid*4+nt)*16 + m16
    const int h0 = (wid * 4) * 16 + m16;
    const __bf16* brow0 = web + (size_t)(h0 +  0) * ENC_;
    const __bf16* brow1 = web + (size_t)(h0 + 16) * ENC_;
    const __bf16* brow2 = web + (size_t)(h0 + 32) * ENC_;
    const __bf16* brow3 = web + (size_t)(h0 + 48) * ENC_;

#define CVT_A(dst, r0, r1, r2, r3) {                                          \
        dst[0]  = (__bf16)r0.x; dst[1]  = (__bf16)r0.y;                       \
        dst[2]  = (__bf16)r0.z; dst[3]  = (__bf16)r0.w;                       \
        dst[4]  = (__bf16)r1.x; dst[5]  = (__bf16)r1.y;                       \
        dst[6]  = (__bf16)r1.z; dst[7]  = (__bf16)r1.w;                       \
        dst[8]  = (__bf16)r2.x; dst[9]  = (__bf16)r2.y;                       \
        dst[10] = (__bf16)r2.z; dst[11] = (__bf16)r2.w;                       \
        dst[12] = (__bf16)r3.x; dst[13] = (__bf16)r3.y;                       \
        dst[14] = (__bf16)r3.z; dst[15] = (__bf16)r3.w; }

#define LOADB(rowptr, dst) {                                                  \
        const v8bf* bp_ = (const v8bf*)((rowptr) + k0 + half * 8);            \
        v8bf lo_ = bp_[0], hi_ = bp_[2];                                      \
        v16bf t_;                                                             \
        for (int i_ = 0; i_ < 8; ++i_) { t_[i_] = lo_[i_]; t_[i_ + 8] = hi_[i_]; } \
        dst = t_; }

    for (int k0 = 0; k0 < ENC_; k0 += 32) {
        // prefetch the next K chunk of both enc rows
        __builtin_prefetch(arow0 + k0 + 32, 0, 1);
        __builtin_prefetch(arow1 + k0 + 32, 0, 1);

        // ---- A fragments: half 0 holds K {0..7,16..23}, half 1 {8..15,24..31}
        const float4* ap0 = (const float4*)(arow0 + k0 + half * 8);
        const float4* ap1 = (const float4*)(arow1 + k0 + half * 8);
        float4 r00 = ap0[0], r01 = ap0[1], r02 = ap0[4], r03 = ap0[5];
        float4 r10 = ap1[0], r11 = ap1[1], r12 = ap1[4], r13 = ap1[5];
        v16bf a0, a1;
        CVT_A(a0, r00, r01, r02, r03);
        CVT_A(a1, r10, r11, r12, r13);

        // ---- B fragments: half 0 holds K {0..7,16..23}, half 1 {8..15,24..31} (row = h)
        v16bf bb;
        LOADB(brow0, bb);
        acc00 = __builtin_amdgcn_wmma_f32_16x16x32_bf16(false, a0, false, bb, (short)0, acc00, false, false);
        acc10 = __builtin_amdgcn_wmma_f32_16x16x32_bf16(false, a1, false, bb, (short)0, acc10, false, false);
        LOADB(brow1, bb);
        acc01 = __builtin_amdgcn_wmma_f32_16x16x32_bf16(false, a0, false, bb, (short)0, acc01, false, false);
        acc11 = __builtin_amdgcn_wmma_f32_16x16x32_bf16(false, a1, false, bb, (short)0, acc11, false, false);
        LOADB(brow2, bb);
        acc02 = __builtin_amdgcn_wmma_f32_16x16x32_bf16(false, a0, false, bb, (short)0, acc02, false, false);
        acc12 = __builtin_amdgcn_wmma_f32_16x16x32_bf16(false, a1, false, bb, (short)0, acc12, false, false);
        LOADB(brow3, bb);
        acc03 = __builtin_amdgcn_wmma_f32_16x16x32_bf16(false, a0, false, bb, (short)0, acc03, false, false);
        acc13 = __builtin_amdgcn_wmma_f32_16x16x32_bf16(false, a1, false, bb, (short)0, acc13, false, false);
    }
#undef LOADB
#undef CVT_A

    // Per-n-tile softmax-input scalars (same for all j of a lane)
    const int   hA = (wid * 4 + 0) * 16 + m16;
    const int   hB = (wid * 4 + 1) * 16 + m16;
    const int   hC = (wid * 4 + 2) * 16 + m16;
    const int   hD = (wid * 4 + 3) * 16 + m16;
    const float swA = sW[b * DEC_ + hA], wvA = Wv[hA];
    const float swB = sW[b * DEC_ + hB], wvB = Wv[hB];
    const float swC = sW[b * DEC_ + hC], wvC = Wv[hC];
    const float swD = sW[b * DEC_ + hD], wvD = Wv[hD];

    // Epilogue: C/D layout — lanes 0-15: N=0-15,M=j ; lanes 16-31: N=0-15,M=8+j
    // Fold the 4 n-tiles (4 different h) inside the lane, then ONE 16-lane
    // xor-reduction per (mt,j) instead of four.
#define EPI(a0v, a1v, a2v, a3v, mt) {                                         \
        for (int j = 0; j < 8; ++j) {                                         \
            float t = wvA * TANHF(swA + a0v[j])                               \
                    + wvB * TANHF(swB + a1v[j])                               \
                    + wvC * TANHF(swC + a2v[j])                               \
                    + wvD * TANHF(swD + a3v[j]);                              \
            RED16(t);                                                         \
            if (m16 == 0) atomicAdd(&lds_atten[(mt) * 16 + j + half * 8], t); \
        } }
    EPI(acc00, acc01, acc02, acc03, 0)
    EPI(acc10, acc11, acc12, acc13, 1)
#undef EPI

    __syncthreads();
    if (tid < 32) atten[m0 + tid] = lds_atten[tid];
}

__global__ __launch_bounds__(256) void k_softmax(const float* __restrict__ atten,
                                                 float* __restrict__ soft) {
    __shared__ float red[256];
    const int b = blockIdx.x, tid = threadIdx.x;
    const float* row = atten + (size_t)b * S_;
    float m = -1e30f;
    for (int i = tid; i < S_; i += 256) m = fmaxf(m, row[i]);
    red[tid] = m; __syncthreads();
    for (int st = 128; st > 0; st >>= 1) {
        if (tid < st) red[tid] = fmaxf(red[tid], red[tid + st]);
        __syncthreads();
    }
    m = red[0]; __syncthreads();
    float sum = 0.f;
    for (int i = tid; i < S_; i += 256) sum += __expf(row[i] - m);
    red[tid] = sum; __syncthreads();
    for (int st = 128; st > 0; st >>= 1) {
        if (tid < st) red[tid] += red[tid + st];
        __syncthreads();
    }
    const float inv = 1.0f / red[0];
    for (int i = tid; i < S_; i += 256) soft[(size_t)b * S_ + i] = __expf(row[i] - m) * inv;
}

__global__ __launch_bounds__(256) void k_context(const float* __restrict__ enc,
                                                 const float* __restrict__ soft,
                                                 float* __restrict__ ctx) {
    const int e4 = threadIdx.x;          // covers e = e4*4 .. e4*4+3
    const int sc = blockIdx.x;           // s chunk 0..7
    const int b  = blockIdx.y;           // 0..31
    const float* sw = soft + (size_t)b * S_;
    float4 acc = {0.f, 0.f, 0.f, 0.f};
    const int s0 = sc * 256;
    for (int s = s0; s < s0 + 256; ++s) {
        float w = sw[s];
        float4 v = ((const float4*)(enc + (size_t)(b * S_ + s) * ENC_))[e4];
        acc.x += w * v.x; acc.y += w * v.y; acc.z += w * v.z; acc.w += w * v.w;
    }
    float* o = ctx + (size_t)b * ENC_ + e4 * 4;
    atomicAdd(o + 0, acc.x); atomicAdd(o + 1, acc.y);
    atomicAdd(o + 2, acc.z); atomicAdd(o + 3, acc.w);
}

extern "C" void kernel_launch(void* const* d_in, const int* in_sizes, int n_in,
                              void* d_out, int out_size, void* d_ws, size_t ws_size,
                              hipStream_t stream) {
    const float* s     = (const float*)d_in[0];
    const float* enc   = (const float*)d_in[1];
    const float* Wattn = (const float*)d_in[2];
    const float* Wv    = (const float*)d_in[3];

    float* out  = (float*)d_out;
    float* ctx  = out;                 // 32*1024 floats
    float* soft = out + B_ * ENC_;     // 32*2048 floats

    char*   ws    = (char*)d_ws;
    __bf16* web   = (__bf16*)ws;                                            // 1 MB
    float*  sW    = (float*)(ws + (size_t)DEC_ * ENC_ * 2);                 // 64 KB
    float*  atten = (float*)(ws + (size_t)DEC_ * ENC_ * 2 + B_ * DEC_ * 4); // 256 KB

    k_convW<<<(DEC_ * ENC_) / 256, 256, 0, stream>>>(Wattn, web);
    k_sW<<<(B_ * DEC_) / 256, 256, 0, stream>>>(s, Wattn, sW);
    k_energy<<<(B_ * S_) / 32, 256, 0, stream>>>(enc, web, sW, Wv, atten);
    k_softmax<<<B_, 256, 0, stream>>>(atten, soft);
    hipMemsetAsync(ctx, 0, (size_t)B_ * ENC_ * sizeof(float), stream);
    k_context<<<dim3(8, B_), 256, 0, stream>>>(enc, soft, ctx);
}
